// ObjectCondensationNet_14723147891229
// MI455X (gfx1250) — matrix operand order repaired and checked by reading, not compiled, not run on hardware
//
#include <hip/hip_runtime.h>
#include <math.h>

#define BB    4
#define NN    2048
#define EE    256
#define HH    8
#define LL    6
#define FFD   1024
#define KNN_K 16
#define ROWS  (BB * NN)        // 8192 token rows
#define QKVC  (3 * EE)         // 768

typedef __bf16 bf16_t;
typedef __attribute__((ext_vector_type(16))) __bf16 bf16x16;
typedef __attribute__((ext_vector_type(8)))  float  f32x8;
typedef int v4i __attribute__((vector_size(16)));

__device__ __forceinline__ int lane_id() { return threadIdx.x & 31; }

// ---------------- CDNA5 async global->LDS copy (ASYNCcnt path) ----------------
#if defined(__HIP_DEVICE_COMPILE__) && __has_builtin(__builtin_amdgcn_global_load_async_to_lds_b128)
#define HAVE_ASYNC_LDS 1
#endif

__device__ __forceinline__ void async_b128(void* l, const void* g) {
#ifdef HAVE_ASYNC_LDS
  __builtin_amdgcn_global_load_async_to_lds_b128(
      (__attribute__((address_space(1))) v4i*)(void*)g,
      (__attribute__((address_space(3))) v4i*)l, 0, 0);
#else
  *(uint4*)l = *(const uint4*)g;
#endif
}

template <int N>
__device__ __forceinline__ void wait_async() {
#ifdef HAVE_ASYNC_LDS
#if __has_builtin(__builtin_amdgcn_s_wait_asynccnt)
  __builtin_amdgcn_s_wait_asynccnt(N);
#else
  asm volatile("s_wait_asynccnt %0" ::"i"(N) : "memory");
#endif
#endif
}

// ---------------- WMMA fragment helpers (bf16, 16x16x32) ----------------
// A fragment: 16(M) x 32(K). Lane r(=lane&15) holds row r; kh=lane>>4.
// VGPR 0..7 : K = kh*16 + 0..15
__device__ __forceinline__ bf16x16 frag_a_rm(const bf16_t* p, int ld) {
  int l = lane_id();
  int r = l & 15, kh = l >> 4;
  const bf16_t* q = p + (size_t)r * ld + kh * 16;
  bf16x16 f;
  ((uint4*)&f)[0] = *(const uint4*)q;          // K = kh*16 .. kh*16+7
  ((uint4*)&f)[1] = *(const uint4*)(q + 8);    // K = kh*16+8 .. kh*16+15
  return f;
}
// B fragment: 32(K) x 16(N); B[k][n] supplied via p = B^T row-major, i.e.
// p[n*ld + k]. Lanes 0-15 hold K=0..15 of col n, lanes 16-31 hold K=16..31.
__device__ __forceinline__ bf16x16 frag_b_rm(const bf16_t* p, int ld) {
  int l = lane_id();
  int n = l & 15, kh = l >> 4;
  const bf16_t* q = p + (size_t)n * ld + kh * 16;
  bf16x16 f;
  ((uint4*)&f)[0] = ((const uint4*)q)[0];
  ((uint4*)&f)[1] = ((const uint4*)q)[1];
  return f;
}
// B fragment from a row-major KxN tile (column gather), used for P @ V.
__device__ __forceinline__ bf16x16 frag_b_cm(const bf16_t* p, int ld) {
  int l = lane_id();
  int n = l & 15, kb = (l >> 4) * 16;
  bf16x16 f;
#pragma unroll
  for (int i = 0; i < 16; ++i) f[i] = p[(size_t)(kb + i) * ld + n];
  return f;
}
__device__ __forceinline__ f32x8 wmma_bf16(bf16x16 a, bf16x16 b, f32x8 c) {
  return __builtin_amdgcn_wmma_f32_16x16x32_bf16(false, a, false, b,
                                                 (short)0, c, false, false);
}

// ---------------- fp32 -> bf16 weight conversion ----------------
__global__ void __launch_bounds__(256)
f32_to_bf16_kernel(const float* __restrict__ in, bf16_t* __restrict__ out, int n) {
  int i = blockIdx.x * 256 + threadIdx.x;
  if (i < n) out[i] = (bf16_t)in[i];
}

// ---------------- KNN geometry ----------------
__global__ void __launch_bounds__(256)
geometry_kernel(const float* __restrict__ hits, float* __restrict__ mean_knn,
                float* __restrict__ density) {
  __shared__ float sx[256], sy[256], sv[256];
  int b = blockIdx.y;
  int n = blockIdx.x * 256 + threadIdx.x;
  const float* hb = hits + (size_t)b * NN * 4;
  float cx = hb[n * 4 + 0], cy = hb[n * 4 + 1], cv = hb[n * 4 + 2];
  float knn[KNN_K];
#pragma unroll
  for (int i = 0; i < KNN_K; ++i) knn[i] = 3.4e38f;
  for (int c0 = 0; c0 < NN; c0 += 256) {
    int t = threadIdx.x;
    sx[t] = hb[(c0 + t) * 4 + 0];
    sy[t] = hb[(c0 + t) * 4 + 1];
    sv[t] = hb[(c0 + t) * 4 + 2];
    __syncthreads();
    for (int m = 0; m < 256; ++m) {
      float dx = cx - sx[m], dy = cy - sy[m];
      float d = sqrtf(dx * dx + dy * dy);
      d = fminf(d, 10000.0f);
      if (cv != sv[m]) d += 10000.0f;
      if (d < knn[KNN_K - 1]) {
        knn[KNN_K - 1] = d;  // one bubble pass places it
#pragma unroll
        for (int j = KNN_K - 1; j > 0; --j) {
          float a = knn[j], bb = knn[j - 1];
          if (a < bb) { knn[j] = bb; knn[j - 1] = a; }
        }
      }
    }
    __syncthreads();
  }
  float s = 0.f;
#pragma unroll
  for (int i = 0; i < KNN_K; ++i) s += knn[i];
  float mn = s * (1.0f / KNN_K);
  mn = fminf(fmaxf(mn, 0.001f), 1000.0f);
  float dn = fminf(fmaxf(1.0f / mn, 0.0f), 1000.0f);
  mean_knn[b * NN + n] = mn;
  density[b * NN + n]  = dn;
}

// ---------------- input embedding (6 -> 256) ----------------
__global__ void __launch_bounds__(256)
input_linear_kernel(const float* __restrict__ hits, const float* __restrict__ mknn,
                    const float* __restrict__ dens, const float* __restrict__ w,
                    const float* __restrict__ bias, float* __restrict__ x) {
  int row = blockIdx.x;
  int e = threadIdx.x;
  const float* hp = hits + (size_t)row * 4;
  float f0 = hp[0], f1 = hp[1], f2 = hp[2], f3 = hp[3];
  float f4 = mknn[row], f5 = dens[row];
  const float* wr = w + e * 6;
  x[(size_t)row * EE + e] =
      f0 * wr[0] + f1 * wr[1] + f2 * wr[2] + f3 * wr[3] + f4 * wr[4] + f5 * wr[5] + bias[e];
}

// ---------------- LayerNorm (one wave per row of 256) ----------------
__global__ void __launch_bounds__(256)
layernorm_kernel(const float* __restrict__ x, const float* __restrict__ w,
                 const float* __restrict__ bta, bf16_t* __restrict__ out_bf,
                 float* __restrict__ out_f) {
  int wave = threadIdx.x >> 5;
  int row = blockIdx.x * 8 + wave;
  int l = lane_id();
  const float* xr = x + (size_t)row * EE;
  float v[8];
  float s = 0.f;
#pragma unroll
  for (int i = 0; i < 8; ++i) { v[i] = xr[l + i * 32]; s += v[i]; }
#pragma unroll
  for (int m = 1; m < 32; m <<= 1) s += __shfl_xor(s, m, 32);
  float mean = s * (1.0f / EE);
  float q = 0.f;
#pragma unroll
  for (int i = 0; i < 8; ++i) { float d = v[i] - mean; q += d * d; }
#pragma unroll
  for (int m = 1; m < 32; m <<= 1) q += __shfl_xor(q, m, 32);
  float rs = rsqrtf(q * (1.0f / EE) + 1e-5f);
#pragma unroll
  for (int i = 0; i < 8; ++i) {
    int e = l + i * 32;
    float o = (v[i] - mean) * rs * w[e] + bta[e];
    out_bf[(size_t)row * EE + e] = (bf16_t)o;
    if (out_f) out_f[(size_t)row * EE + e] = o;
  }
}

// ---------------- WMMA GEMM: C[M,N] = A[M,K] @ W[N,K]^T + bias ----------------
// 8 waves, tile 128(M) x 64(N). LDS double-buffered, async global->LDS staging.
// Per k-step: A tile 128x32 (8KB), W tile 64x32 (4KB); 3 async b128 per thread.
__global__ void __launch_bounds__(256)
gemm_wmma_kernel(const bf16_t* __restrict__ A, const bf16_t* __restrict__ W,
                 const float* __restrict__ bias, const float* __restrict__ Res,
                 float* __restrict__ Cf, bf16_t* __restrict__ Cb,
                 int M, int Nout, int Kdim, int relu) {
  __shared__ __align__(16) bf16_t sA[2][128 * 32];
  __shared__ __align__(16) bf16_t sW[2][64 * 32];
  int tid = threadIdx.x;
  int wave = tid >> 5;
  int mblk = blockIdx.y * 128;
  int nblk = blockIdx.x * 64;

  // staging indices
  int ar = tid >> 1;             // A row 0..127
  int ac = (tid & 1) * 16;       // A col offset (halfs)
  int wr = tid >> 2;             // W row 0..63
  int wc = (tid & 3) * 8;        // W col offset (halfs)
  const bf16_t* gA = A + (size_t)(mblk + ar) * Kdim + ac;
  const bf16_t* gW = W + (size_t)(nblk + wr) * Kdim + wc;

  f32x8 acc[4] = {{}, {}, {}, {}};
  int steps = Kdim >> 5;

  // prologue: stage tile 0
  {
    bf16_t* la = &sA[0][ar * 32 + ac];
    async_b128(la, gA);
    async_b128(la + 8, gA + 8);
    async_b128(&sW[0][wr * 32 + wc], gW);
  }
  for (int i = 0; i < steps; ++i) {
    int cur = i & 1;
    if (i + 1 < steps) {
      int nxt = cur ^ 1;
      int k0 = (i + 1) << 5;
      bf16_t* la = &sA[nxt][ar * 32 + ac];
      async_b128(la, gA + k0);
      async_b128(la + 8, gA + k0 + 8);
      async_b128(&sW[nxt][wr * 32 + wc], gW + k0);
      wait_async<3>();   // tile i complete (in-order); tile i+1 may be in flight
    } else {
      wait_async<0>();
    }
    __syncthreads();
    bf16x16 a = frag_a_rm(&sA[cur][wave * 16 * 32], 32);
#pragma unroll
    for (int t = 0; t < 4; ++t) {
      bf16x16 b = frag_b_rm(&sW[cur][t * 16 * 32], 32);
      acc[t] = wmma_bf16(a, b, acc[t]);
    }
    __syncthreads();  // readers done before this buffer is refilled
  }

  int l = lane_id();
  int co = l & 15, ro = (l >> 4) * 8;
#pragma unroll
  for (int t = 0; t < 4; ++t) {
#pragma unroll
    for (int j = 0; j < 8; ++j) {
      int row = mblk + wave * 16 + ro + j;
      int col = nblk + t * 16 + co;
      float v = acc[t][j] + bias[col];
      if (Res)  v += Res[(size_t)row * Nout + col];
      if (relu) v = fmaxf(v, 0.0f);
      if (Cf) Cf[(size_t)row * Nout + col] = v;
      if (Cb) Cb[(size_t)row * Nout + col] = (bf16_t)v;
    }
  }
}

// ---------------- Flash attention (H=8, D=32), WMMA both matmuls ----------------
// qkv: bf16 [B*N, 768] with q|k|v at col offsets 0/256/512, head at h*32.
// out: bf16 [B*N, 256].
__global__ void __launch_bounds__(128)
attention_kernel(const bf16_t* __restrict__ qkv, bf16_t* __restrict__ out) {
  __shared__ __align__(16) bf16_t kT[32 * 32];
  __shared__ __align__(16) bf16_t vT[32 * 32];
  __shared__ __align__(16) bf16_t pS[4][16 * 32];
  int b = blockIdx.z, h = blockIdx.y;
  int wave = threadIdx.x >> 5;
  int q0 = blockIdx.x * 64 + wave * 16;
  const bf16_t* qptr = qkv + (size_t)(b * NN + q0) * QKVC + h * 32;
  bf16x16 aq = frag_a_rm(qptr, QKVC);
  f32x8 o0 = {}, o1 = {};
  float mj[8], lj[8];
#pragma unroll
  for (int j = 0; j < 8; ++j) { mj[j] = -1e30f; lj[j] = 0.0f; }
  const float scale = 0.17677669529663687f;  // 1/sqrt(32)
  int tid = threadIdx.x;
  int lrow = tid >> 2;           // 0..31 key row within chunk
  int lgrp = (tid & 3) * 8;      // 8-half group
  int l = lane_id();
  int co = l & 15, ro = (l >> 4) * 8;

  for (int m0 = 0; m0 < NN; m0 += 32) {
    const bf16_t* kr = qkv + (size_t)(b * NN + m0 + lrow) * QKVC + EE + h * 32 + lgrp;
    async_b128(&kT[lrow * 32 + lgrp], kr);
    async_b128(&vT[lrow * 32 + lgrp], kr + EE);
    wait_async<0>();
    __syncthreads();

    f32x8 z = {};
    f32x8 s0 = wmma_bf16(aq, frag_b_rm(&kT[0], 32), z);        // keys m0..m0+15
    f32x8 s1 = wmma_bf16(aq, frag_b_rm(&kT[16 * 32], 32), z);  // keys m0+16..m0+31
#pragma unroll
    for (int j = 0; j < 8; ++j) {
      float v0 = s0[j] * scale, v1 = s1[j] * scale;
      float t = fmaxf(v0, v1);
      t = fmaxf(t, __shfl_xor(t, 1, 32));
      t = fmaxf(t, __shfl_xor(t, 2, 32));
      t = fmaxf(t, __shfl_xor(t, 4, 32));
      t = fmaxf(t, __shfl_xor(t, 8, 32));
      float mn = fmaxf(mj[j], t);
      float corr = __expf(mj[j] - mn);
      float p0 = __expf(v0 - mn);
      float p1 = __expf(v1 - mn);
      float rsum = p0 + p1;
      rsum += __shfl_xor(rsum, 1, 32);
      rsum += __shfl_xor(rsum, 2, 32);
      rsum += __shfl_xor(rsum, 4, 32);
      rsum += __shfl_xor(rsum, 8, 32);
      lj[j] = lj[j] * corr + rsum;
      mj[j] = mn;
      o0[j] *= corr;
      o1[j] *= corr;
      pS[wave][(ro + j) * 32 + co]      = (bf16_t)p0;
      pS[wave][(ro + j) * 32 + 16 + co] = (bf16_t)p1;
    }
    __syncthreads();
    bf16x16 ap = frag_a_rm(&pS[wave][0], 32);            // P tile 16x32
    o0 = wmma_bf16(ap, frag_b_cm(&vT[0], 32), o0);       // dims 0..15
    o1 = wmma_bf16(ap, frag_b_cm(&vT[16], 32), o1);      // dims 16..31
    __syncthreads();
  }
#pragma unroll
  for (int j = 0; j < 8; ++j) {
    int q = q0 + ro + j;
    float inv = 1.0f / lj[j];
    size_t base = (size_t)(b * NN + q) * EE + h * 32;
    out[base + co]      = (bf16_t)(o0[j] * inv);
    out[base + 16 + co] = (bf16_t)(o1[j] * inv);
  }
}

// ---------------- beta head (E -> 1), one wave per row ----------------
__global__ void __launch_bounds__(256)
beta_kernel(const float* __restrict__ xn, const float* __restrict__ bw,
            const float* __restrict__ bb, float* __restrict__ out) {
  int wave = threadIdx.x >> 5;
  int row = blockIdx.x * 8 + wave;
  int l = lane_id();
  const float* xr = xn + (size_t)row * EE;
  float s = 0.f;
#pragma unroll
  for (int i = 0; i < 8; ++i) s += xr[l + i * 32] * bw[l + i * 32];
#pragma unroll
  for (int m = 1; m < 32; m <<= 1) s += __shfl_xor(s, m, 32);
  if (l == 0) out[row] = s + bb[0];
}

// ---------------- host launcher ----------------
extern "C" void kernel_launch(void* const* d_in, const int* in_sizes, int n_in,
                              void* d_out, int out_size, void* d_ws, size_t ws_size,
                              hipStream_t stream) {
  const float* hits   = (const float*)d_in[0];
  const float* inp_w  = (const float*)d_in[1];
  const float* inp_b  = (const float*)d_in[2];
  const float* ln1_w  = (const float*)d_in[3];
  const float* ln1_b  = (const float*)d_in[4];
  const float* qkv_w  = (const float*)d_in[5];
  const float* qkv_b  = (const float*)d_in[6];
  const float* o_w    = (const float*)d_in[7];
  const float* o_b    = (const float*)d_in[8];
  const float* ln2_w  = (const float*)d_in[9];
  const float* ln2_b  = (const float*)d_in[10];
  const float* ff1_w  = (const float*)d_in[11];
  const float* ff1_b  = (const float*)d_in[12];
  const float* ff2_w  = (const float*)d_in[13];
  const float* ff2_b  = (const float*)d_in[14];
  const float* fn_w   = (const float*)d_in[15];
  const float* fn_b   = (const float*)d_in[16];
  const float* beta_w = (const float*)d_in[17];
  const float* beta_b = (const float*)d_in[18];
  const float* emb_w  = (const float*)d_in[19];
  const float* emb_b  = (const float*)d_in[20];
  float* out = (float*)d_out;

  char* wsb = (char*)d_ws;
  size_t off = 0;
  auto alloc = [&](size_t bytes) -> void* {
    void* p = wsb + off;
    off += (bytes + 255) & ~(size_t)255;
    return p;
  };
  float*  x       = (float*)alloc((size_t)ROWS * EE * 4);
  float*  xn_f    = (float*)alloc((size_t)ROWS * EE * 4);
  bf16_t* xn_bf   = (bf16_t*)alloc((size_t)ROWS * EE * 2);
  bf16_t* qkv_bf  = (bf16_t*)alloc((size_t)ROWS * QKVC * 2);
  bf16_t* attn_bf = (bf16_t*)alloc((size_t)ROWS * EE * 2);
  bf16_t* h_bf    = (bf16_t*)alloc((size_t)ROWS * FFD * 2);
  float*  mknn    = (float*)alloc((size_t)ROWS * 4);
  float*  dens    = (float*)alloc((size_t)ROWS * 4);
  bf16_t* qkvw_bf = (bf16_t*)alloc((size_t)LL * QKVC * EE * 2);
  bf16_t* ow_bf   = (bf16_t*)alloc((size_t)LL * EE * EE * 2);
  bf16_t* ff1w_bf = (bf16_t*)alloc((size_t)LL * FFD * EE * 2);
  bf16_t* ff2w_bf = (bf16_t*)alloc((size_t)LL * EE * FFD * 2);
  bf16_t* embw_bf = (bf16_t*)alloc((size_t)EE * EE * 2);

  auto cvt = [&](const float* src, bf16_t* dst, int n) {
    f32_to_bf16_kernel<<<(n + 255) / 256, 256, 0, stream>>>(src, dst, n);
  };
  cvt(qkv_w, qkvw_bf, LL * QKVC * EE);
  cvt(o_w,   ow_bf,   LL * EE * EE);
  cvt(ff1_w, ff1w_bf, LL * FFD * EE);
  cvt(ff2_w, ff2w_bf, LL * EE * FFD);
  cvt(emb_w, embw_bf, EE * EE);

  geometry_kernel<<<dim3(NN / 256, BB), 256, 0, stream>>>(hits, mknn, dens);
  input_linear_kernel<<<ROWS, 256, 0, stream>>>(hits, mknn, dens, inp_w, inp_b, x);

  for (int li = 0; li < LL; ++li) {
    layernorm_kernel<<<ROWS / 8, 256, 0, stream>>>(x, ln1_w + li * EE, ln1_b + li * EE,
                                                   xn_bf, nullptr);
    gemm_wmma_kernel<<<dim3(QKVC / 64, ROWS / 128), 256, 0, stream>>>(
        xn_bf, qkvw_bf + (size_t)li * QKVC * EE, qkv_b + li * QKVC,
        nullptr, nullptr, qkv_bf, ROWS, QKVC, EE, 0);
    attention_kernel<<<dim3(NN / 64, HH, BB), 128, 0, stream>>>(qkv_bf, attn_bf);
    gemm_wmma_kernel<<<dim3(EE / 64, ROWS / 128), 256, 0, stream>>>(
        attn_bf, ow_bf + (size_t)li * EE * EE, o_b + li * EE,
        x, x, nullptr, ROWS, EE, EE, 0);
    layernorm_kernel<<<ROWS / 8, 256, 0, stream>>>(x, ln2_w + li * EE, ln2_b + li * EE,
                                                   xn_bf, nullptr);
    gemm_wmma_kernel<<<dim3(FFD / 64, ROWS / 128), 256, 0, stream>>>(
        xn_bf, ff1w_bf + (size_t)li * FFD * EE, ff1_b + li * FFD,
        nullptr, nullptr, h_bf, ROWS, FFD, EE, 1);
    gemm_wmma_kernel<<<dim3(EE / 64, ROWS / 128), 256, 0, stream>>>(
        h_bf, ff2w_bf + (size_t)li * EE * FFD, ff2_b + li * EE,
        x, x, nullptr, ROWS, EE, FFD, 0);
  }

  layernorm_kernel<<<ROWS / 8, 256, 0, stream>>>(x, fn_w, fn_b, xn_bf, xn_f);
  beta_kernel<<<ROWS / 8, 256, 0, stream>>>(xn_f, beta_w, beta_b, out);
  gemm_wmma_kernel<<<dim3(EE / 64, ROWS / 128), 256, 0, stream>>>(
      xn_bf, embw_bf, emb_b, nullptr, out + ROWS, nullptr, ROWS, EE, EE, 0);
}
